// MaskedLSTM_33466385170734
// MI455X (gfx1250) — matrix operand order repaired and checked by reading, not compiled, not run on hardware
//
#include <hip/hip_runtime.h>
#include <hip/hip_bf16.h>

typedef __attribute__((ext_vector_type(16))) __bf16 v16bf;
typedef __attribute__((ext_vector_type(8)))  __bf16 v8bf;
typedef __attribute__((ext_vector_type(8)))  float  v8f;
typedef int v4i_ __attribute__((vector_size(16)));
typedef v4i_ __attribute__((address_space(1)))* g4p_;   // global int4*
typedef v4i_ __attribute__((address_space(3)))* l4p_;   // LDS int4*

#define B_ 64
#define S_ 256
#define I_ 1024
#define H_ 1024
#define APAD_ 1032   // padded LDS row stride (elements): 2064B -> rows land on distinct bank groups

#if defined(__has_builtin)
#  if __has_builtin(__builtin_amdgcn_global_load_async_to_lds_b128)
#    define HAVE_ASYNC_LDS 1
#  endif
#endif
#ifndef HAVE_ASYNC_LDS
#  define HAVE_ASYNC_LDS 0
#endif

__device__ __forceinline__ unsigned short f32_to_bf16_rne(float f) {
    unsigned int u = __float_as_uint(f);
    u += 0x7FFFu + ((u >> 16) & 1u);   // round-to-nearest-even
    return (unsigned short)(u >> 16);
}

__device__ __forceinline__ float sigmoidf_(float x) {
    return 1.0f / (1.0f + __expf(-x));
}

__device__ __forceinline__ v16bf make_afrag(const unsigned short* base, int kk, int half) {
    v8bf alo = *reinterpret_cast<const v8bf*>(base + kk + half * 16);
    v8bf ahi = *reinterpret_cast<const v8bf*>(base + kk + half * 16 + 8);
    return __builtin_shufflevector(alo, ahi, 0,1,2,3,4,5,6,7,8,9,10,11,12,13,14,15);
}

// ---------------------------------------------------------------- converters
__global__ void cvt_f32_bf16(const float* __restrict__ src,
                             unsigned short* __restrict__ dst, int n) {
    int i = blockIdx.x * blockDim.x + threadIdx.x;
    if (i < n) dst[i] = f32_to_bf16_rne(src[i]);
}

__global__ void init_state(float* __restrict__ c,
                           unsigned short* __restrict__ h0,
                           unsigned* __restrict__ counter, int n) {
    int i = blockIdx.x * blockDim.x + threadIdx.x;
    if (i < n) { c[i] = 0.0f; h0[i] = 0; }
    if (i == 0) *counter = 0u;
}

// --------------------------------------------------- phase 1: input masked GEMMs
// Wave = (gate, nt) over a 4-M-tile stripe: 8 accumulators (W,Wm x 4 row-tiles),
// so each weight B-frag feeds 4 WMMA and weights are read once per 64-row group.
__global__ void __launch_bounds__(256)
input_mlin_gemm(const unsigned short* __restrict__ xbf,   // (B*S, I) bf16
                const unsigned short* __restrict__ wbf,   // 8 mats of (H,I) bf16
                float* __restrict__ gates) {              // (4,S,B,H) f32
    const int lane = threadIdx.x & 31;
    const int half = lane >> 4;
    const int ln16 = lane & 15;
    const int wi   = blockIdx.x * 8 + (threadIdx.x >> 5); // 0..65535
    const int mgrp = wi >> 8;     // 0..255  (64-row group)
    const int rem  = wi & 255;
    const int gate = rem >> 6;    // 0..3
    const int nt   = rem & 63;    // 0..63

    const size_t MATSZ = (size_t)H_ * I_;
    const int col = nt * 16 + ln16;
    const unsigned short* wrowW = wbf + (size_t)(2 * gate)     * MATSZ + (size_t)col * I_;
    const unsigned short* wrowM = wbf + (size_t)(2 * gate + 1) * MATSZ + (size_t)col * I_;
    const unsigned short* arow[4];
    #pragma unroll
    for (int m = 0; m < 4; ++m)
        arow[m] = xbf + (size_t)((mgrp * 4 + m) * 16 + ln16) * I_;

    v8f accW[4] = {};
    v8f accM[4] = {};
    for (int kk = 0; kk < I_; kk += 32) {
        v16bf bw = *reinterpret_cast<const v16bf*>(wrowW + kk + half * 16);
        v16bf bm = *reinterpret_cast<const v16bf*>(wrowM + kk + half * 16);
        // locality 3 -> WGP-scope prefetch: warm the near cache for the B stream
        __builtin_prefetch((const void*)(wrowW + kk + 512), 0, 3);
        __builtin_prefetch((const void*)(wrowM + kk + 512), 0, 3);
        #pragma unroll
        for (int m = 0; m < 4; ++m) {
            v16bf a = make_afrag(arow[m], kk, half);
            accW[m] = __builtin_amdgcn_wmma_f32_16x16x32_bf16(false, a, false, bw,
                                                              (short)0, accW[m], false, false);
            accM[m] = __builtin_amdgcn_wmma_f32_16x16x32_bf16(false, a, false, bm,
                                                              (short)0, accM[m], false, false);
        }
    }

    const size_t GST = (size_t)S_ * B_ * H_;
    #pragma unroll
    for (int m = 0; m < 4; ++m) {
        #pragma unroll
        for (int v = 0; v < 8; ++v) {
            int r = (mgrp * 4 + m) * 16 + v + 8 * half;   // row in (B*S)
            int b = r >> 8;                               // r / S_
            int s = r & 255;                              // r % S_
            size_t base = ((size_t)s * B_ + b) * H_ + col;
            gates[(size_t)gate * GST + base] = accW[m][v] * sigmoidf_(accM[m][v]);
        }
    }
}

// --------------------------------------------------- phase 2: persistent recurrence
// 64 blocks x 256 threads (8 waves). Block = one 16-column stripe (nt); wave g
// computes matrix g (U_f,Um_f,U_i,Um_i,U_o,Um_o,U_c,Um_c) over all 4 M-tiles, so
// each U weight byte is read exactly once per step (L2-resident, 16MB/step).
// h is staged into LDS per step via async global->LDS copies; grid-wide
// monotonic-counter barrier between steps.
__global__ void __launch_bounds__(256)
lstm_recurrence(const unsigned short* __restrict__ ubf,   // 8 mats of (H,H) bf16
                const float* __restrict__ gates,          // (4,S,B,H) f32
                const float* __restrict__ b_f, const float* __restrict__ b_i,
                const float* __restrict__ b_o, const float* __restrict__ b_c,
                float* __restrict__ cbuf,                 // (B,H) f32
                unsigned short* __restrict__ hbf0,        // (B,H) bf16 ping
                unsigned short* __restrict__ hbf1,        // (B,H) bf16 pong
                float* __restrict__ out,                  // (B,S,H) f32
                float* __restrict__ out_h, float* __restrict__ out_c,
                unsigned* __restrict__ counter) {
    __shared__ unsigned short astage[64 * APAD_];   // whole h, padded rows (~129KB)
    __shared__ float xch[8 * 4 * 256];              // 8 mats x 4 M-tiles x 16x16 f32 (32KB)

    const int lane = threadIdx.x & 31;
    const int half = lane >> 4;
    const int ln16 = lane & 15;
    const int g    = threadIdx.x >> 5;   // wave = matrix index 0..7
    const int nt   = blockIdx.x;         // 0..63

    const int j = nt * 16 + ln16;
    const unsigned short* urow = ubf + (size_t)g * H_ * H_ + (size_t)j * H_;
    const unsigned short* larow[4];
    #pragma unroll
    for (int m = 0; m < 4; ++m)
        larow[m] = &astage[(m * 16 + ln16) * APAD_];

    const int e  = threadIdx.x;          // epilogue element-in-tile 0..255
    const int ej = nt * 16 + (e & 15);
    const float bfv = b_f[ej], biv = b_i[ej], bov = b_o[ej], bcv = b_c[ej];
    const size_t GST = (size_t)S_ * B_ * H_;
    const unsigned nb = gridDim.x;

    for (int t = 0; t < S_; ++t) {
        const unsigned short* hin  = (t & 1) ? hbf1 : hbf0;
        unsigned short*       hout = (t & 1) ? hbf0 : hbf1;

        // ---- stage h (64x1024 bf16) into LDS: 8192 16B chunks over 256 threads
        for (int i = 0; i < 32; ++i) {
            int c   = i * 256 + threadIdx.x;
            int row = c >> 7;             // 0..63
            int off = (c & 127) * 8;      // element offset within row
            const unsigned short* gp = hin + (size_t)row * H_ + off;
            unsigned short*       lp = &astage[row * APAD_ + off];
#if HAVE_ASYNC_LDS
            __builtin_amdgcn_global_load_async_to_lds_b128((g4p_)gp, (l4p_)lp, 0, 0);
#else
            *reinterpret_cast<v8bf*>(lp) = *reinterpret_cast<const v8bf*>(gp);
#endif
        }
#if HAVE_ASYNC_LDS
#  if __has_builtin(__builtin_amdgcn_s_wait_asynccnt)
        __builtin_amdgcn_s_wait_asynccnt(0);
#  else
        asm volatile("s_wait_asynccnt 0" ::: "memory");
#  endif
#endif
        __syncthreads();

        // ---- K-loop: stream B from L2 once, A-frags from LDS, 4 WMMA per B-frag
        v8f acc[4] = {};
        for (int kk = 0; kk < H_; kk += 32) {
            v16bf bfr = *reinterpret_cast<const v16bf*>(urow + kk + half * 16);
            __builtin_prefetch((const void*)(urow + kk + 512), 0, 3);
            #pragma unroll
            for (int m = 0; m < 4; ++m) {
                v16bf a = make_afrag(larow[m], kk, half);
                acc[m] = __builtin_amdgcn_wmma_f32_16x16x32_bf16(false, a, false, bfr,
                                                                 (short)0, acc[m], false, false);
            }
        }

        // ---- exchange the 8x4 accumulator tiles through LDS
        #pragma unroll
        for (int m = 0; m < 4; ++m)
            #pragma unroll
            for (int v = 0; v < 8; ++v)
                xch[(g * 4 + m) * 256 + (v + 8 * half) * 16 + ln16] = acc[m][v];
        __syncthreads();

        // ---- fused masked-linear + LSTM cell epilogue: 4 elements per thread
        #pragma unroll
        for (int m = 0; m < 4; ++m) {
            int eb = m * 16 + (e >> 4);   // batch index 0..63
            float aw0 = xch[(0 * 4 + m) * 256 + e], am0 = xch[(1 * 4 + m) * 256 + e];
            float aw1 = xch[(2 * 4 + m) * 256 + e], am1 = xch[(3 * 4 + m) * 256 + e];
            float aw2 = xch[(4 * 4 + m) * 256 + e], am2 = xch[(5 * 4 + m) * 256 + e];
            float aw3 = xch[(6 * 4 + m) * 256 + e], am3 = xch[(7 * 4 + m) * 256 + e];

            size_t gidx = ((size_t)t * B_ + eb) * H_ + ej;
            float xf = gates[gidx];
            float xi = gates[GST + gidx];
            float xo = gates[2 * GST + gidx];
            float xc = gates[3 * GST + gidx];

            float f  = sigmoidf_(xf + sigmoidf_(am0) * aw0 + bfv);
            float ii = sigmoidf_(xi + sigmoidf_(am1) * aw1 + biv);
            float o  = sigmoidf_(xo + sigmoidf_(am2) * aw2 + bov);
            float cd = tanhf(    xc + sigmoidf_(am3) * aw3 + bcv);

            size_t hidx = (size_t)eb * H_ + ej;
            float cn = cd * ii + f * cbuf[hidx];
            float hn = o * cn;
            cbuf[hidx] = cn;
            hout[hidx] = f32_to_bf16_rne(hn);
            out[((size_t)eb * S_ + t) * H_ + ej] = hn;
            if (t == S_ - 1) { out_h[hidx] = hn; out_c[hidx] = cn; }
        }

        // ---- make h/c visible device-wide, then grid-wide barrier
        __threadfence();
        __syncthreads();
        if (threadIdx.x == 0) {
            __hip_atomic_fetch_add(counter, 1u, __ATOMIC_ACQ_REL,
                                   __HIP_MEMORY_SCOPE_AGENT);
            const unsigned target = (unsigned)(t + 1) * nb;
            while (__hip_atomic_load(counter, __ATOMIC_ACQUIRE,
                                     __HIP_MEMORY_SCOPE_AGENT) < target)
                __builtin_amdgcn_s_sleep(1);
        }
        __syncthreads();
    }
}

// ------------------------------------------------------------------ launcher
extern "C" void kernel_launch(void* const* d_in, const int* in_sizes, int n_in,
                              void* d_out, int out_size, void* d_ws, size_t ws_size,
                              hipStream_t stream) {
    (void)in_sizes; (void)n_in; (void)out_size; (void)ws_size;
    const float* x = (const float*)d_in[0];

    char* ws = (char*)d_ws;
    size_t off = 0;
    float*          gates = (float*)(ws + off);          off += (size_t)4 * S_ * B_ * H_ * sizeof(float);
    unsigned short* wbf   = (unsigned short*)(ws + off); off += (size_t)8 * H_ * I_ * 2;
    unsigned short* ubf   = (unsigned short*)(ws + off); off += (size_t)8 * H_ * H_ * 2;
    unsigned short* xbf   = (unsigned short*)(ws + off); off += (size_t)B_ * S_ * I_ * 2;
    float*          cbuf  = (float*)(ws + off);          off += (size_t)B_ * H_ * sizeof(float);
    unsigned short* hbf0  = (unsigned short*)(ws + off); off += (size_t)B_ * H_ * 2;
    unsigned short* hbf1  = (unsigned short*)(ws + off); off += (size_t)B_ * H_ * 2;
    unsigned*       ctr   = (unsigned*)(ws + off);       off += 256;

    const int nmat = H_ * I_;  // == H_*H_
    for (int m = 0; m < 8; ++m)   // W_f, Wm_f, W_i, Wm_i, W_o, Wm_o, W_c, Wm_c
        cvt_f32_bf16<<<(nmat + 255) / 256, 256, 0, stream>>>(
            (const float*)d_in[1 + m], wbf + (size_t)m * nmat, nmat);
    for (int m = 0; m < 8; ++m)   // U_f, Um_f, U_i, Um_i, U_o, Um_o, U_c, Um_c
        cvt_f32_bf16<<<(nmat + 255) / 256, 256, 0, stream>>>(
            (const float*)d_in[9 + m], ubf + (size_t)m * nmat, nmat);

    const int nx = B_ * S_ * I_;
    cvt_f32_bf16<<<(nx + 255) / 256, 256, 0, stream>>>(x, xbf, nx);

    const int nh = B_ * H_;
    init_state<<<(nh + 255) / 256, 256, 0, stream>>>(cbuf, hbf0, ctr, nh);

    // 65536 wave-stripes / 8 waves per block
    input_mlin_gemm<<<8192, 256, 0, stream>>>(xbf, wbf, gates);

    float* out   = (float*)d_out;
    float* out_h = out + (size_t)B_ * S_ * H_;
    float* out_c = out_h + (size_t)B_ * H_;
    lstm_recurrence<<<64, 256, 0, stream>>>(ubf, gates,
        (const float*)d_in[17], (const float*)d_in[18],
        (const float*)d_in[19], (const float*)d_in[20],
        cbuf, hbf0, hbf1, out, out_h, out_c, ctr);
}
